// DeformableTransformer_2302102471071
// MI455X (gfx1250) — compile-verified, matrix operand order checked by reading, not run on hardware
//
#include <hip/hip_runtime.h>
#include <hip/hip_bf16.h>
#include <math.h>

// ---------------- problem constants (from reference) ----------------
#define Bsz   8
#define NQT   350
#define NP    300
#define NTQ   50      // track queries = 350-300
#define Dm    256
#define Hh    8
#define LVLc  4
#define Pc    4
#define NLc   6
#define H2c   2
#define DH2   128
#define DFFc  1024
#define S_TOT 20197   // 100*152 + 50*76 + 25*38 + 13*19

typedef __bf16 bf16;
typedef bf16  v16bf __attribute__((ext_vector_type(16)));
typedef bf16  v8bf  __attribute__((ext_vector_type(8)));
typedef bf16  v4bf  __attribute__((ext_vector_type(4)));
typedef float v8f   __attribute__((ext_vector_type(8)));

// =====================================================================
// Generic tiled WMMA GEMM:  C[z] = act(A[z] @ B[z] + bias)   (f32 in/out)
//  - A: [M,K] f32 row-major (lda), per-z offset z0*sA0 + z1*sA1, z0=bz/nz1, z1=bz%nz1
//  - B: TRANSB ? [N,K] : [K,N] f32 row-major (ldb)
//  - f32 -> bf16 conversion while staging tiles into LDS, float4 global loads
//  - block tile 128x64, 8 waves; wave w owns rows [16w,16w+16), 4 N-subtiles
//  - inner op: v_wmma_f32_16x16x32_bf16
//  Requirements: K % 32 == 0, all operands 16B aligned, lda/ldb % 4 == 0.
// =====================================================================
template<bool TRANSB, bool RELU>
__global__ __launch_bounds__(256) void gemm_wmma_k(
    const float* __restrict__ A, int lda, long long sA0, long long sA1,
    const float* __restrict__ B, int ldb, long long sB0, long long sB1,
    const float* __restrict__ bias,
    float* __restrict__ C, int ldc, long long sC0, long long sC1,
    int M, int N, int K, int nz1)
{
  __shared__ __align__(16) bf16 As[128 * 32];
  __shared__ __align__(16) bf16 Bs[64 * 32];

  const int bz = blockIdx.z;
  const int z0 = bz / nz1, z1 = bz - z0 * nz1;
  A += z0 * sA0 + z1 * sA1;
  B += z0 * sB0 + z1 * sB1;
  C += z0 * sC0 + z1 * sC1;

  const int m0 = blockIdx.y * 128;
  const int n0 = blockIdx.x * 64;
  const int wv = threadIdx.x >> 5;
  const int lane = threadIdx.x & 31;

  v8f acc[4];
  const v8f vzero = {0.f, 0.f, 0.f, 0.f, 0.f, 0.f, 0.f, 0.f};
#pragma unroll
  for (int i = 0; i < 4; ++i) acc[i] = vzero;

  // ISA 16-bit A-matrix 16x32 layout: lane<16 -> K {0..7,16..23}; lane>=16 -> +8
  const int mrow = lane & 15;
  const int koff = (lane >> 4) << 3;

  for (int k0 = 0; k0 < K; k0 += 32) {
    // ---- stage A tile (128x32): 1024 float4 slots, 4 per thread ----
#pragma unroll
    for (int i = 0; i < 4; ++i) {
      int slot = i * 256 + threadIdx.x;
      int r = slot >> 3;
      int c4 = (slot & 7) << 2;
      int gm = m0 + r;
      float4 v = {0.f, 0.f, 0.f, 0.f};
      if (gm < M) {
        const float* p = A + (long long)gm * lda + k0 + c4;
        v = *(const float4*)p;
        if (i == 0 && k0 + 32 < K) __builtin_prefetch(p + 32, 0, 1);
      }
      v4bf bv;
      bv[0] = (bf16)v.x; bv[1] = (bf16)v.y; bv[2] = (bf16)v.z; bv[3] = (bf16)v.w;
      *(v4bf*)(As + r * 32 + c4) = bv;
    }
    // ---- stage B tile transposed into Bs[n][k] (64x32): 512 slots, 2/thread ----
    if (TRANSB) {
      // B is [N,K]: rows are contiguous in K -> direct vector copy
#pragma unroll
      for (int i = 0; i < 2; ++i) {
        int slot = i * 256 + threadIdx.x;
        int n = slot >> 3;
        int c4 = (slot & 7) << 2;
        int gn = n0 + n;
        float4 v = {0.f, 0.f, 0.f, 0.f};
        if (gn < N) v = *(const float4*)(B + (long long)gn * ldb + k0 + c4);
        v4bf bv;
        bv[0] = (bf16)v.x; bv[1] = (bf16)v.y; bv[2] = (bf16)v.z; bv[3] = (bf16)v.w;
        *(v4bf*)(Bs + n * 32 + c4) = bv;
      }
    } else {
      // B is [K,N]: load float4 along N (coalesced), scatter into transposed tile
#pragma unroll
      for (int i = 0; i < 2; ++i) {
        int slot = i * 256 + threadIdx.x;
        int k = slot >> 4;
        int nb = (slot & 15) << 2;
        int gn0 = n0 + nb;
        float4 v = {0.f, 0.f, 0.f, 0.f};
        if (gn0 + 3 < N) {
          v = *(const float4*)(B + (long long)(k0 + k) * ldb + gn0);
        } else {
          const float* p = B + (long long)(k0 + k) * ldb;
          v.x = (gn0 + 0 < N) ? p[gn0 + 0] : 0.f;
          v.y = (gn0 + 1 < N) ? p[gn0 + 1] : 0.f;
          v.z = (gn0 + 2 < N) ? p[gn0 + 2] : 0.f;
          v.w = (gn0 + 3 < N) ? p[gn0 + 3] : 0.f;
        }
        Bs[(nb + 0) * 32 + k] = (bf16)v.x;
        Bs[(nb + 1) * 32 + k] = (bf16)v.y;
        Bs[(nb + 2) * 32 + k] = (bf16)v.z;
        Bs[(nb + 3) * 32 + k] = (bf16)v.w;
      }
    }
    __syncthreads();

    const bf16* arow = &As[(wv * 16 + mrow) * 32];
    v8bf alo = *(const v8bf*)(arow + koff);
    v8bf ahi = *(const v8bf*)(arow + koff + 16);
    v16bf afrag = __builtin_shufflevector(alo, ahi,
        0, 1, 2, 3, 4, 5, 6, 7, 8, 9, 10, 11, 12, 13, 14, 15);

#pragma unroll
    for (int nt = 0; nt < 4; ++nt) {
      const bf16* brow = &Bs[(nt * 16 + mrow) * 32];
      v8bf blo = *(const v8bf*)(brow + koff);
      v8bf bhi = *(const v8bf*)(brow + koff + 16);
      v16bf bfrag = __builtin_shufflevector(blo, bhi,
          0, 1, 2, 3, 4, 5, 6, 7, 8, 9, 10, 11, 12, 13, 14, 15);
      acc[nt] = __builtin_amdgcn_wmma_f32_16x16x32_bf16(
          false, afrag, false, bfrag, (short)0, acc[nt], false, false);
    }
    __syncthreads();
  }

  // C/D layout: VGPR r, lanes 0-15 -> M=r, N=lane ; lanes 16-31 -> M=8+r
  const int mb = m0 + wv * 16 + ((lane >> 4) << 3);
  const int nbl = lane & 15;
#pragma unroll
  for (int nt = 0; nt < 4; ++nt) {
    int n = n0 + nt * 16 + nbl;
    if (n >= N) continue;
    float bvv = bias ? bias[n] : 0.f;
#pragma unroll
    for (int r = 0; r < 8; ++r) {
      int m = mb + r;
      if (m < M) {
        float v = acc[nt][r] + bvv;
        if (RELU) v = fmaxf(v, 0.f);
        C[(long long)m * ldc + n] = v;
      }
    }
  }
}

// =====================================================================
// Specialized value-projection GEMM: C(bf16)[M,256] = A(bf16)[M,256] @ W(f32)[256,256] + bias
// A already bf16 -> pure b128 copies into LDS (no converts in hot loop).
// =====================================================================
__global__ __launch_bounds__(256) void gemm_vproj_k(
    const bf16* __restrict__ A, const float* __restrict__ W,
    const float* __restrict__ bias, bf16* __restrict__ C, int M)
{
  __shared__ __align__(16) bf16 As[128 * 32];
  __shared__ __align__(16) bf16 Bs[64 * 32];

  const int m0 = blockIdx.y * 128;
  const int n0 = blockIdx.x * 64;
  const int wv = threadIdx.x >> 5;
  const int lane = threadIdx.x & 31;

  v8f acc[4];
  const v8f vzero = {0.f, 0.f, 0.f, 0.f, 0.f, 0.f, 0.f, 0.f};
#pragma unroll
  for (int i = 0; i < 4; ++i) acc[i] = vzero;

  const int mrow = lane & 15;
  const int koff = (lane >> 4) << 3;
  const v8bf bzero = {(bf16)0.f, (bf16)0.f, (bf16)0.f, (bf16)0.f,
                      (bf16)0.f, (bf16)0.f, (bf16)0.f, (bf16)0.f};

  for (int k0 = 0; k0 < 256; k0 += 32) {
    // ---- stage A tile (128x32 bf16): 512 v8bf slots, 2 per thread ----
#pragma unroll
    for (int i = 0; i < 2; ++i) {
      int slot = i * 256 + threadIdx.x;
      int r = slot >> 2;
      int c8 = (slot & 3) << 3;
      int gm = m0 + r;
      v8bf v = bzero;
      if (gm < M) {
        const bf16* p = A + (long long)gm * 256 + k0 + c8;
        v = *(const v8bf*)p;
        if (i == 0 && k0 < 224) __builtin_prefetch(p + 32, 0, 1);
      }
      *(v8bf*)(As + r * 32 + c8) = v;
    }
    // ---- stage W tile transposed (f32 [K,N] -> Bs[n][k]); N=256 so no guards ----
#pragma unroll
    for (int i = 0; i < 2; ++i) {
      int slot = i * 256 + threadIdx.x;
      int k = slot >> 4;
      int nb = (slot & 15) << 2;
      float4 v = *(const float4*)(W + (long long)(k0 + k) * 256 + n0 + nb);
      Bs[(nb + 0) * 32 + k] = (bf16)v.x;
      Bs[(nb + 1) * 32 + k] = (bf16)v.y;
      Bs[(nb + 2) * 32 + k] = (bf16)v.z;
      Bs[(nb + 3) * 32 + k] = (bf16)v.w;
    }
    __syncthreads();

    const bf16* arow = &As[(wv * 16 + mrow) * 32];
    v8bf alo = *(const v8bf*)(arow + koff);
    v8bf ahi = *(const v8bf*)(arow + koff + 16);
    v16bf afrag = __builtin_shufflevector(alo, ahi,
        0, 1, 2, 3, 4, 5, 6, 7, 8, 9, 10, 11, 12, 13, 14, 15);

#pragma unroll
    for (int nt = 0; nt < 4; ++nt) {
      const bf16* brow = &Bs[(nt * 16 + mrow) * 32];
      v8bf blo = *(const v8bf*)(brow + koff);
      v8bf bhi = *(const v8bf*)(brow + koff + 16);
      v16bf bfrag = __builtin_shufflevector(blo, bhi,
          0, 1, 2, 3, 4, 5, 6, 7, 8, 9, 10, 11, 12, 13, 14, 15);
      acc[nt] = __builtin_amdgcn_wmma_f32_16x16x32_bf16(
          false, afrag, false, bfrag, (short)0, acc[nt], false, false);
    }
    __syncthreads();
  }

  const int mb = m0 + wv * 16 + ((lane >> 4) << 3);
  const int nbl = lane & 15;
#pragma unroll
  for (int nt = 0; nt < 4; ++nt) {
    int n = n0 + nt * 16 + nbl;
    float bvv = bias[n];
#pragma unroll
    for (int r = 0; r < 8; ++r) {
      int m = mb + r;
      if (m < M) C[(long long)m * 256 + n] = (bf16)(acc[nt][r] + bvv);
    }
  }
}

// ---------------- elementwise / small kernels ----------------

// one-time f32 -> bf16 downconvert (vectorized float4 -> v4bf)
__global__ void f32_to_bf16_k(const float* __restrict__ in, bf16* __restrict__ out, int n4) {
  int i = blockIdx.x * 256 + threadIdx.x;
  if (i >= n4) return;
  float4 v = ((const float4*)in)[i];
  v4bf b;
  b[0] = (bf16)v.x; b[1] = (bf16)v.y; b[2] = (bf16)v.z; b[3] = (bf16)v.w;
  ((v4bf*)out)[i] = b;
}

__global__ void copy_prop_k(const float* __restrict__ tgt, float* __restrict__ out) {
  int i = blockIdx.x * 256 + threadIdx.x;
  if (i >= Bsz * NP * Dm) return;
  int b = i / (NP * Dm);
  int rem = i - b * (NP * Dm);
  out[i] = tgt[(long long)b * NQT * Dm + rem];
}

__global__ void copy_pref_k(const float* __restrict__ refp, float* __restrict__ out) {
  int i = blockIdx.x * 256 + threadIdx.x;
  if (i >= Bsz * NP * 4) return;
  int b = i / (NP * 4);
  int rem = i - b * (NP * 4);
  out[i] = refp[(long long)b * NQT * 4 + rem];
}

__global__ void add_k(const float* __restrict__ a, const float* __restrict__ b,
                      float* __restrict__ c, int n) {
  int i = blockIdx.x * 256 + threadIdx.x;
  if (i < n) c[i] = a[i] + b[i];
}

// pos2posemb: out[(b,q), c*64 + 2j(+1)] = sin/cos(pos_c * 2pi / 10000^(j/32))
__global__ void posemb_k(const float* __restrict__ refp, float* __restrict__ out) {
  int bq = blockIdx.x;
  int b = bq / NP, q = bq - b * NP;
  int t = threadIdx.x;
  int c = t >> 6, r = t & 63, j = r >> 1;
  float tj = __powf(10000.f, (float)j * (1.f / 32.f));
  float v = refp[((long long)b * NQT + q) * 4 + c] * 6.28318530717958647f / tj;
  out[(long long)bq * Dm + t] = (r & 1) ? __cosf(v) : __sinf(v);
}

// softmax over LVL*P=16 for each (b,nq,h) row inside [B,NP,128]
__global__ void softmax16_k(float* __restrict__ a, int nrows) {
  int i = blockIdx.x * 256 + threadIdx.x;
  if (i >= nrows) return;
  float* p = a + (long long)(i >> 3) * 128 + (long long)(i & 7) * 16;
  float m = -1e30f;
#pragma unroll
  for (int j = 0; j < 16; ++j) m = fmaxf(m, p[j]);
  float e[16], s = 0.f;
#pragma unroll
  for (int j = 0; j < 16; ++j) { e[j] = __expf(p[j] - m); s += e[j]; }
  float inv = 1.f / s;
#pragma unroll
  for (int j = 0; j < 16; ++j) p[j] = e[j] * inv;
}

// deformable bilinear sampling + attention-weighted sum (value in bf16)
// block = (b,q); 256 threads = (h in 0..7) x (dh channel 0..31) == flat channel t
__global__ void deform_sample_k(const bf16* __restrict__ value,
                                const float* __restrict__ offs,
                                const float* __restrict__ attnw,
                                const float* __restrict__ refp,
                                const float* __restrict__ vrat,
                                float* __restrict__ out) {
  const int HL[4] = {100, 50, 25, 13};
  const int WL[4] = {152, 76, 38, 19};
  const int ST[4] = {0, 15200, 19000, 19950};

  int bq = blockIdx.x;
  int b = bq / NP, q = bq - b * NP;
  int t = threadIdx.x;
  int h = t >> 5;

  const float* rp = refp + ((long long)b * NQT + q) * 4;
  float r0 = rp[0], r1 = rp[1], r2 = rp[2], r3 = rp[3];

  float acc = 0.f;
#pragma unroll
  for (int l = 0; l < 4; ++l) {
    float svx = vrat[((long long)b * LVLc + l) * 2 + 0];
    float svy = vrat[((long long)b * LVLc + l) * 2 + 1];
    float rx = r0 * svx, ry = r1 * svy, rw = r2 * svx, rh = r3 * svy;
    float wl = (float)WL[l], hl = (float)HL[l];
#pragma unroll
    for (int p = 0; p < 4; ++p) {
      int ch = (h * 4 + l) * 4 + p;
      long long ob = ((long long)bq * Dm) + ch * 2;
      float ox = offs[ob], oy = offs[ob + 1];
      float aw = attnw[(long long)bq * 128 + ch];
      // loc = ref_xy + off/P * ref_wh * 0.5
      float lx = rx + ox * 0.125f * rw;
      float ly = ry + oy * 0.125f * rh;
      float x = lx * wl - 0.5f, y = ly * hl - 0.5f;
      float x0 = floorf(x), y0 = floorf(y);
      float fx = x - x0, fy = y - y0;
#pragma unroll
      for (int dy = 0; dy < 2; ++dy) {
#pragma unroll
        for (int dx = 0; dx < 2; ++dx) {
          float xi = x0 + dx, yi = y0 + dy;
          float w = (dx ? fx : 1.f - fx) * (dy ? fy : 1.f - fy);
          bool valid = (xi >= 0.f) && (xi <= wl - 1.f) && (yi >= 0.f) && (yi <= hl - 1.f);
          float xc = fminf(fmaxf(xi, 0.f), wl - 1.f);
          float yc = fminf(fmaxf(yi, 0.f), hl - 1.f);
          int idx = ST[l] + (int)yc * WL[l] + (int)xc;
          float v = (float)value[((long long)b * S_TOT + idx) * Dm + t];
          acc += aw * w * (valid ? v : 0.f);
        }
      }
    }
  }
  out[(long long)bq * Dm + t] = acc;
}

// x = LayerNorm(x + add) * w + b, one block per 256-wide row
__global__ void add_ln_k(float* __restrict__ x, const float* __restrict__ add,
                         const float* __restrict__ w, const float* __restrict__ bb) {
  long long row = blockIdx.x;
  int t = threadIdx.x;
  float v = x[row * Dm + t] + add[row * Dm + t];
  __shared__ float red[256];
  red[t] = v; __syncthreads();
  for (int s = 128; s > 0; s >>= 1) { if (t < s) red[t] += red[t + s]; __syncthreads(); }
  float mean = red[0] * (1.f / 256.f);
  __syncthreads();
  float d = v - mean;
  red[t] = d * d; __syncthreads();
  for (int s = 128; s > 0; s >>= 1) { if (t < s) red[t] += red[t + s]; __syncthreads(); }
  float var = red[0] * (1.f / 256.f);
  x[row * Dm + t] = d * rsqrtf(var + 1e-5f) * w[t] + bb[t];
}

// per-head L2 norm over 128 channels; blockIdx = row*2 + h, row = b*nr + r
__global__ void rownorm_k(const float* __restrict__ x, float* __restrict__ out, int nr) {
  int bh = blockIdx.x;
  int row = bh >> 1, h = bh & 1;
  int b = row / nr, r = row - b * nr;
  const float* base = x + (long long)row * Dm + h * DH2;
  int t = threadIdx.x;  // 128 threads
  float v = base[t];
  __shared__ float red[128];
  red[t] = v * v; __syncthreads();
  for (int s = 64; s > 0; s >>= 1) { if (t < s) red[t] += red[t + s]; __syncthreads(); }
  if (t == 0) out[(long long)(b * 2 + h) * nr + r] = sqrtf(red[0]);
}

// softmax(s/sqrt(dh)) and cosine epilogue; blockIdx = (b*2+h)*300 + qi
__global__ void attn_post_k(const float* __restrict__ S, const float* __restrict__ qn,
                            const float* __restrict__ kn, float* __restrict__ ocos,
                            float* __restrict__ omm, int nk) {
  int row = blockIdx.x;
  int bh = row / NP;
  const float* s = S + (long long)row * nk;
  float q = qn[row];
  const float* knb = kn + (long long)bh * nk;
  int t = threadIdx.x;
  const float scale = 0.0883883476483184405f;  // 1/sqrt(128)
  __shared__ float red[256];

  float lm = -1e30f;
  for (int j = t; j < nk; j += 256) lm = fmaxf(lm, s[j] * scale);
  red[t] = lm; __syncthreads();
  for (int st = 128; st > 0; st >>= 1) { if (t < st) red[t] = fmaxf(red[t], red[t + st]); __syncthreads(); }
  float rowmax = red[0];
  __syncthreads();

  float ls = 0.f;
  for (int j = t; j < nk; j += 256) ls += __expf(s[j] * scale - rowmax);
  red[t] = ls; __syncthreads();
  for (int st = 128; st > 0; st >>= 1) { if (t < st) red[t] += red[t + st]; __syncthreads(); }
  float denom = red[0];

  for (int j = t; j < nk; j += 256) {
    float sv = s[j];
    omm[(long long)row * nk + j] = __expf(sv * scale - rowmax) / denom;
    ocos[(long long)row * nk + j] = sv / ((q + 1e-6f) * (knb[j] + 1e-6f));
  }
}

// =====================================================================
// host orchestration
// =====================================================================
extern "C" void kernel_launch(void* const* d_in, const int* in_sizes, int n_in,
                              void* d_out, int out_size, void* d_ws, size_t ws_size,
                              hipStream_t stream) {
  const float* tgt  = (const float*)d_in[0];
  const float* refp = (const float*)d_in[1];
  const float* src  = (const float*)d_in[2];
  const float* vrat = (const float*)d_in[5];
  const float* Woff = (const float*)d_in[7];
  const float* boff = (const float*)d_in[8];
  const float* Wa   = (const float*)d_in[9];
  const float* ba   = (const float*)d_in[10];
  const float* Wv   = (const float*)d_in[11];
  const float* bv   = (const float*)d_in[12];
  const float* Wout = (const float*)d_in[13];
  const float* bout = (const float*)d_in[14];
  const float* Wl1  = (const float*)d_in[15];
  const float* bl1  = (const float*)d_in[16];
  const float* Wl2  = (const float*)d_in[17];
  const float* bl2  = (const float*)d_in[18];
  const float* Wq   = (const float*)d_in[19];
  const float* bq   = (const float*)d_in[20];
  const float* Wk   = (const float*)d_in[21];
  const float* bk   = (const float*)d_in[22];
  const float* n1w  = (const float*)d_in[23];
  const float* n1b  = (const float*)d_in[24];
  const float* n3w  = (const float*)d_in[25];
  const float* n3b  = (const float*)d_in[26];

  float* out = (float*)d_out;
  float* o_prop  = out;                 // [8,300,256]
  float* o_pref  = out + 614400;        // [8,300,4]
  float* o_ptcos = out + 624000;        // [8,2,300,50]
  float* o_ppcos = out + 864000;        // [8,2,300,300]
  float* o_ptmm  = out + 2304000;       // [8,2,300,50]
  float* o_ppmm  = out + 2544000;       // [8,2,300,300]

  char* wsb = (char*)d_ws;
  size_t cur = 0;
  auto allocb = [&](size_t bytes) -> void* {
    void* p = wsb + cur;
    cur += (bytes + 255) & ~(size_t)255;
    return p;
  };
  const long long NVALE = (long long)Bsz * S_TOT * Dm;       // 41,363,456 elements
  bf16*  srcbf   = (bf16*)allocb((size_t)NVALE * 2);
  bf16*  valbf   = (bf16*)allocb((size_t)NVALE * 2);
  float* ppos    = (float*)allocb((size_t)Bsz * NP * Dm * 4);
  float* qbuf    = (float*)allocb((size_t)Bsz * NP * Dm * 4);
  float* offb    = (float*)allocb((size_t)Bsz * NP * Dm * 4);
  float* attnw   = (float*)allocb((size_t)Bsz * NP * 128 * 4);
  float* ca_pre  = (float*)allocb((size_t)Bsz * NP * Dm * 4);
  float* ca_proj = (float*)allocb((size_t)Bsz * NP * Dm * 4);
  float* ffm     = (float*)allocb((size_t)Bsz * NP * DFFc * 4);
  float* ffo     = (float*)allocb((size_t)Bsz * NP * Dm * 4);
  float* qp      = (float*)allocb((size_t)Bsz * NP * Dm * 4);
  float* ktb     = (float*)allocb((size_t)Bsz * NTQ * Dm * 4);
  float* kpb     = (float*)allocb((size_t)Bsz * NP * Dm * 4);
  float* spt     = (float*)allocb((size_t)Bsz * H2c * NP * NTQ * 4);
  float* spp     = (float*)allocb((size_t)Bsz * H2c * NP * NP * 4);
  float* qnorm   = (float*)allocb((size_t)Bsz * H2c * NP * 4);
  float* ktnorm  = (float*)allocb((size_t)Bsz * H2c * NTQ * 4);
  float* kpnorm  = (float*)allocb((size_t)Bsz * H2c * NP * 4);

  auto gemm = [&](const float* A, int lda, long long sA0, long long sA1,
                  const float* Bp, int ldb, long long sB0, long long sB1,
                  const float* bias, float* Cp, int ldc, long long sC0, long long sC1,
                  int M, int N, int K, int nz1, int Z, bool transb, bool relu) {
    dim3 g((N + 63) / 64, (M + 127) / 128, Z);
    if (transb)
      gemm_wmma_k<true, false><<<g, 256, 0, stream>>>(A, lda, sA0, sA1, Bp, ldb, sB0, sB1,
                                                      bias, Cp, ldc, sC0, sC1, M, N, K, nz1);
    else if (relu)
      gemm_wmma_k<false, true><<<g, 256, 0, stream>>>(A, lda, sA0, sA1, Bp, ldb, sB0, sB1,
                                                      bias, Cp, ldc, sC0, sC1, M, N, K, nz1);
    else
      gemm_wmma_k<false, false><<<g, 256, 0, stream>>>(A, lda, sA0, sA1, Bp, ldb, sB0, sB1,
                                                       bias, Cp, ldc, sC0, sC1, M, N, K, nz1);
  };

  const int NROW = Bsz * NP;              // 2400
  const int NVAL = Bsz * S_TOT;           // 161576

  copy_prop_k<<<(Bsz * NP * Dm + 255) / 256, 256, 0, stream>>>(tgt, o_prop);
  copy_pref_k<<<(Bsz * NP * 4 + 255) / 256, 256, 0, stream>>>(refp, o_pref);
  posemb_k<<<NROW, 256, 0, stream>>>(refp, ppos);
  // one-time src -> bf16 precache (src is reused by all 6 layers)
  {
    int n4 = (int)(NVALE / 4);
    f32_to_bf16_k<<<(n4 + 255) / 256, 256, 0, stream>>>(src, srcbf, n4);
  }

  for (int l = 0; l < NLc; ++l) {
    const float* Woff_l = Woff + (size_t)l * Dm * 256;
    const float* boff_l = boff + (size_t)l * 256;
    const float* Wa_l   = Wa   + (size_t)l * Dm * 128;
    const float* ba_l   = ba   + (size_t)l * 128;
    const float* Wv_l   = Wv   + (size_t)l * Dm * Dm;
    const float* bv_l   = bv   + (size_t)l * Dm;
    const float* Wout_l = Wout + (size_t)l * Dm * Dm;
    const float* bout_l = bout + (size_t)l * Dm;
    const float* Wl1_l  = Wl1  + (size_t)l * Dm * DFFc;
    const float* bl1_l  = bl1  + (size_t)l * DFFc;
    const float* Wl2_l  = Wl2  + (size_t)l * DFFc * Dm;
    const float* bl2_l  = bl2  + (size_t)l * Dm;

    // q = proposals + p_pos
    add_k<<<(NROW * Dm + 255) / 256, 256, 0, stream>>>(o_prop, ppos, qbuf, NROW * Dm);
    // sampling offsets and attention logits
    gemm(qbuf, Dm, 0, 0, Woff_l, 256, 0, 0, boff_l, offb, 256, 0, 0, NROW, 256, Dm, 1, 1, false, false);
    gemm(qbuf, Dm, 0, 0, Wa_l, 128, 0, 0, ba_l, attnw, 128, 0, 0, NROW, 128, Dm, 1, 1, false, false);
    softmax16_k<<<(NROW * Hh + 255) / 256, 256, 0, stream>>>(attnw, NROW * Hh);
    // value projection (dominant GEMM): bf16 in, bf16 out
    {
      dim3 g(4, (NVAL + 127) / 128, 1);
      gemm_vproj_k<<<g, 256, 0, stream>>>(srcbf, Wv_l, bv_l, valbf, NVAL);
    }
    // deformable sampling (bf16 gather)
    deform_sample_k<<<NROW, 256, 0, stream>>>(valbf, offb, attnw, refp, vrat, ca_pre);
    // output projection + residual LN
    gemm(ca_pre, Dm, 0, 0, Wout_l, Dm, 0, 0, bout_l, ca_proj, Dm, 0, 0, NROW, Dm, Dm, 1, 1, false, false);
    add_ln_k<<<NROW, 256, 0, stream>>>(o_prop, ca_proj, n1w + (size_t)l * Dm, n1b + (size_t)l * Dm);
    // FFN
    gemm(o_prop, Dm, 0, 0, Wl1_l, DFFc, 0, 0, bl1_l, ffm, DFFc, 0, 0, NROW, DFFc, Dm, 1, 1, false, true);
    gemm(ffm, DFFc, 0, 0, Wl2_l, Dm, 0, 0, bl2_l, ffo, Dm, 0, 0, NROW, Dm, DFFc, 1, 1, false, false);
    add_ln_k<<<NROW, 256, 0, stream>>>(o_prop, ffo, n3w + (size_t)l * Dm, n3b + (size_t)l * Dm);
  }

  // weight_attn only matters for the final layer (earlier results are overwritten)
  const int l = NLc - 1;
  const float* Wq_l = Wq + (size_t)l * Dm * Dm;
  const float* bq_l = bq + (size_t)l * Dm;
  const float* Wk_l = Wk + (size_t)l * Dm * Dm;
  const float* bk_l = bk + (size_t)l * Dm;

  gemm(o_prop, Dm, 0, 0, Wq_l, Dm, 0, 0, bq_l, qp, Dm, 0, 0, NROW, Dm, Dm, 1, 1, false, false);
  // track queries: tgt[:,300:350] -> per-batch strided A, batched over z=B
  gemm(tgt + (size_t)NP * Dm, Dm, (long long)NQT * Dm, 0, Wk_l, Dm, 0, 0, bk_l,
       ktb, Dm, (long long)NTQ * Dm, 0, NTQ, Dm, Dm, 1, Bsz, false, false);
  gemm(o_prop, Dm, 0, 0, Wk_l, Dm, 0, 0, bk_l, kpb, Dm, 0, 0, NROW, Dm, Dm, 1, 1, false, false);

  rownorm_k<<<NROW * 2, 128, 0, stream>>>(qp, qnorm, NP);
  rownorm_k<<<Bsz * NTQ * 2, 128, 0, stream>>>(ktb, ktnorm, NTQ);
  rownorm_k<<<NROW * 2, 128, 0, stream>>>(kpb, kpnorm, NP);

  // scores: per (b,h) NT-GEMM  [300x128] @ [nk x128]^T, z = b*2+h
  gemm(qp, Dm, (long long)NP * Dm, DH2, ktb, Dm, (long long)NTQ * Dm, DH2, nullptr,
       spt, NTQ, (long long)H2c * NP * NTQ, (long long)NP * NTQ,
       NP, NTQ, DH2, 2, Bsz * H2c, true, false);
  gemm(qp, Dm, (long long)NP * Dm, DH2, kpb, Dm, (long long)NP * Dm, DH2, nullptr,
       spp, NP, (long long)H2c * NP * NP, (long long)NP * NP,
       NP, NP, DH2, 2, Bsz * H2c, true, false);

  attn_post_k<<<Bsz * H2c * NP, 256, 0, stream>>>(spt, qnorm, ktnorm, o_ptcos, o_ptmm, NTQ);
  attn_post_k<<<Bsz * H2c * NP, 256, 0, stream>>>(spp, qnorm, kpnorm, o_ppcos, o_ppmm, NP);
}
